// ImprovedMambaCore_31078383354560
// MI455X (gfx1250) — matrix operand order repaired and checked by reading, not compiled, not run on hardware
//
#include <hip/hip_runtime.h>
#include <math.h>

// ---------------------------------------------------------------------------
// Mamba-style block for MI455X (gfx1250, wave32, WMMA).
// Heavy GEMMs use v_wmma_f32_16x16x32_bf16 (bf16 in, f32 accumulate).
// A-tiles staged global->LDS via global_load_async_to_lds_b128 (ASYNCcnt).
// ---------------------------------------------------------------------------

typedef __attribute__((ext_vector_type(16))) __bf16 v16bf;
typedef __attribute__((ext_vector_type(8)))  float  v8f;
typedef __attribute__((ext_vector_type(4)))  int    v4i;

#define HID   768
#define STATE 16
#define KW    4
#define INNER 1536
#define BSZ   2
#define LEN   1024
#define NTOK  (BSZ * LEN)          // 2048
#define KIM   (KW * INNER)         // 6144 im2col K

#if defined(__AMDGCN__) && __has_builtin(__builtin_amdgcn_global_load_async_to_lds_b128) && \
    __has_builtin(__builtin_amdgcn_s_wait_asynccnt)
#define USE_ASYNC_COPY 1
#else
#define USE_ASYNC_COPY 0
#endif

union AFrag { v16bf v; unsigned int u[8]; };
union CFrag { v8f   v; float f[8]; };

__device__ __forceinline__ float silu_f(float v)     { return v / (1.0f + __expf(-v)); }
__device__ __forceinline__ float softplus_f(float v) { return (v > 20.0f) ? v : log1pf(__expf(v)); }

// ---------------------------------------------------------------------------
// f32 -> bf16 conversion (weights / activations)
// ---------------------------------------------------------------------------
__global__ __launch_bounds__(256) void cvt_f32_bf16(const float* __restrict__ src,
                                                    __bf16* __restrict__ dst, long n) {
    long i = (long)blockIdx.x * 256 + threadIdx.x;
    if (i < n) dst[i] = (__bf16)src[i];
}

// ---------------------------------------------------------------------------
// im2col for the K=4 "same"-padded channel-mixing conv.
// row t = [ x_inner[t-1] | x_inner[t] | x_inner[t+1] | x_inner[t+2] ],
// zeros across batch boundaries.  x_inner = first INNER columns of xz.
// ---------------------------------------------------------------------------
__global__ __launch_bounds__(256) void pack_im2col(const float* __restrict__ xz,
                                                   __bf16* __restrict__ im) {
    long idx = (long)blockIdx.x * 256 + threadIdx.x;
    if (idx >= (long)NTOK * KIM) return;
    int t = (int)(idx / KIM);
    int r = (int)(idx - (long)t * KIM);
    int k = r / INNER;
    int c = r - k * INNER;
    int b = t >> 10;
    int l = t & 1023;
    int lp = l + k - 1;                       // pad_left = (K-1)/2 = 1
    float v = 0.0f;
    if (lp >= 0 && lp < LEN)
        v = xz[((size_t)(b * LEN + lp)) * (2 * INNER) + c];
    im[idx] = (__bf16)v;
}

// ---------------------------------------------------------------------------
// WMMA bf16 GEMM:  C[M,N] (f32) = A[M,K] * B[K,N], A/B bf16 row-major.
// Block: 256 threads = 8 waves.  Block tile 64x64; each wave owns a 16x32
// slice (two 16x16 WMMA tiles sharing one A fragment).
// act: 0 = none, 1 = SiLU, 2 = softplus.  Optional bias[N] and bf16 mirror Cb.
// Requires M%64==0, K%32==0, N%32==0 (N<64 handled by uniform guards).
// ---------------------------------------------------------------------------
__global__ __launch_bounds__(256) void gemm_bf16_wmma(const __bf16* __restrict__ A,
                                                      const __bf16* __restrict__ B,
                                                      float* __restrict__ C,
                                                      __bf16* __restrict__ Cb,
                                                      const float* __restrict__ bias,
                                                      int M, int N, int K, int act) {
    __shared__ __align__(16) __bf16 As[64][32];   // [m][k]
    __shared__ __align__(16) __bf16 Bs[64][32];   // transposed: [n][k]

    const int tid  = threadIdx.x;
    const int lane = tid & 31;
    const int wv   = tid >> 5;            // wave 0..7
    const int tr   = wv >> 1;             // wave tile row 0..3  (x16)
    const int tc   = wv & 1;              // wave tile col pair 0..1 (x32)
    const int m0   = blockIdx.y * 64;
    const int n0   = blockIdx.x * 64;
    const int half = lane >> 4;           // 0 / 1
    const int mr   = lane & 15;           // row (A/D) or col (B/D) within tile

    CFrag acc0, acc1;
#pragma unroll
    for (int i = 0; i < 8; ++i) { acc0.f[i] = 0.0f; acc1.f[i] = 0.0f; }

    for (int kk = 0; kk < K; kk += 32) {
        __syncthreads();
        // ---- stage A tile 64x32 (straight copy; async path if available) ----
#if USE_ASYNC_COPY
        {
            int row = tid >> 2;               // 0..63
            int seg = (tid & 3) * 8;          // 0,8,16,24 (bf16 elems = 16B)
            const __bf16* gp = A + (size_t)(m0 + row) * K + kk + seg;
            __builtin_amdgcn_global_load_async_to_lds_b128(
                (__attribute__((address_space(1))) v4i*)(size_t)gp,
                (__attribute__((address_space(3))) v4i*)(unsigned int)(size_t)&As[row][seg],
                0, 0);
        }
#else
#pragma unroll
        for (int i = 0; i < 4; ++i) {
            int id  = tid + 256 * i;          // 0..1023 dwords
            int row = id >> 4;
            int cu  = id & 15;
            unsigned int val = *(const unsigned int*)(A + (size_t)(m0 + row) * K + kk + cu * 2);
            *(unsigned int*)&As[row][cu * 2] = val;
        }
#endif
        // ---- stage B tile 32x64 transposed into [n][k] ----
        {
            int k  = tid >> 3;                // 0..31
            int n  = (tid & 7) * 8;           // 0..56
            int gn = n0 + n;
            if (gn < N) {
                const __bf16* src = B + (size_t)(kk + k) * N + gn;
#pragma unroll
                for (int i = 0; i < 8; ++i) Bs[n + i][k] = src[i];
            }
        }
        // ---- prefetch next K-step tiles ----
        if (kk + 32 < K) {
            __builtin_prefetch(A + (size_t)(m0 + (tid >> 2)) * K + kk + 32, 0, 1);
            if (n0 < N)
                __builtin_prefetch(B + (size_t)(kk + 32 + (tid >> 3)) * N + n0, 0, 1);
        }
#if USE_ASYNC_COPY
        __builtin_amdgcn_s_wait_asynccnt(0);
#endif
        __syncthreads();

        // ---- fragments per ISA 16-bit layouts (b128-mergeable reads) ----
        AFrag af, bf0, bf1;
#pragma unroll
        for (int i = 0; i < 4; ++i)          // A: K = 8h + {0..7}
            af.u[i] = *(const unsigned int*)&As[tr * 16 + mr][half * 8 + 2 * i];
#pragma unroll
        for (int i = 0; i < 4; ++i)          // A: K = 16 + 8h + {0..7}
            af.u[4 + i] = *(const unsigned int*)&As[tr * 16 + mr][16 + half * 8 + 2 * i];
#pragma unroll
        for (int i = 0; i < 8; ++i)          // B: K = 8h + {0..7}, 16 + 8h + {0..7}, col = mr
            bf0.u[i] = *(const unsigned int*)&Bs[tc * 32 + mr][(i >> 2) * 16 + half * 8 + 2 * (i & 3)];
#pragma unroll
        for (int i = 0; i < 8; ++i)
            bf1.u[i] = *(const unsigned int*)&Bs[tc * 32 + 16 + mr][(i >> 2) * 16 + half * 8 + 2 * (i & 3)];

        acc0.v = __builtin_amdgcn_wmma_f32_16x16x32_bf16(
            false, af.v, false, bf0.v, (short)0, acc0.v, false, false);
        acc1.v = __builtin_amdgcn_wmma_f32_16x16x32_bf16(
            false, af.v, false, bf1.v, (short)0, acc1.v, false, false);
    }

    // ---- epilogue: D element r -> (row 8h+r, col mr) ----
    const int mBase = m0 + tr * 16 + half * 8;
#pragma unroll
    for (int j = 0; j < 2; ++j) {
        const CFrag& acc = j ? acc1 : acc0;
        int n = n0 + tc * 32 + j * 16 + mr;
        if (n < N) {
            float bv = bias ? bias[n] : 0.0f;
#pragma unroll
            for (int r = 0; r < 8; ++r) {
                float v = acc.f[r] + bv;
                if (act == 1)      v = silu_f(v);
                else if (act == 2) v = softplus_f(v);
                C[(size_t)(mBase + r) * N + n] = v;
                if (Cb) Cb[(size_t)(mBase + r) * N + n] = (__bf16)v;
            }
        }
    }
}

// ---------------------------------------------------------------------------
// Selective-scan: one thread per (batch, channel); 16-wide state in registers.
// B_t / C_t (32 floats per step) broadcast through LDS.
// ---------------------------------------------------------------------------
__global__ __launch_bounds__(256) void scan_kernel(const float* __restrict__ xconv,
                                                   const float* __restrict__ delta,
                                                   const float* __restrict__ BC,
                                                   const float* __restrict__ A_log,
                                                   float* __restrict__ y) {
    const int d = blockIdx.x * 256 + threadIdx.x;   // channel 0..1535
    const int b = blockIdx.y;
    __shared__ float Bsh[STATE];
    __shared__ float Csh[STATE];

    float Av[STATE];
#pragma unroll
    for (int s = 0; s < STATE; ++s) Av[s] = -__expf(A_log[(size_t)d * STATE + s]);
    float h[STATE];
#pragma unroll
    for (int s = 0; s < STATE; ++s) h[s] = 0.0f;

    for (int t = 0; t < LEN; ++t) {
        const size_t row = (size_t)(b * LEN + t);
        __syncthreads();
        if (threadIdx.x < 2 * STATE) {
            float v = BC[row * (2 * STATE) + threadIdx.x];
            if (threadIdx.x < STATE) Bsh[threadIdx.x] = v;
            else                     Csh[threadIdx.x - STATE] = v;
        }
        __syncthreads();
        float dt = delta[row * INNER + d];
        float dx = dt * xconv[row * INNER + d];
        float acc = 0.0f;
#pragma unroll
        for (int s = 0; s < STATE; ++s) {
            h[s] = __expf(dt * Av[s]) * h[s] + dx * Bsh[s];
            acc += h[s] * Csh[s];
        }
        y[row * INNER + d] = acc;
    }
}

// ---------------------------------------------------------------------------
// y = (y_scan + D * x_conv) * silu(z)  -> bf16 for out_proj GEMM
// ---------------------------------------------------------------------------
__global__ __launch_bounds__(256) void gate_kernel(const float* __restrict__ ys,
                                                   const float* __restrict__ xconv,
                                                   const float* __restrict__ xz,
                                                   const float* __restrict__ Dp,
                                                   __bf16* __restrict__ yb) {
    long idx = (long)blockIdx.x * 256 + threadIdx.x;
    if (idx >= (long)NTOK * INNER) return;
    int t = (int)(idx / INNER);
    int d = (int)(idx - (long)t * INNER);
    float z   = xz[(size_t)t * (2 * INNER) + INNER + d];
    float val = (ys[idx] + Dp[d] * xconv[idx]) * silu_f(z);
    yb[idx] = (__bf16)val;
}

// ---------------------------------------------------------------------------
// LayerNorm over last dim (768); one block per token, 3 elems/thread.
// ---------------------------------------------------------------------------
__global__ __launch_bounds__(256) void layernorm_kernel(const float* __restrict__ in,
                                                        const float* __restrict__ gamma,
                                                        const float* __restrict__ beta,
                                                        float* __restrict__ out) {
    const int t = blockIdx.x;
    const int tid = threadIdx.x;
    __shared__ float red[256];

    float v[3];
    float s = 0.0f;
#pragma unroll
    for (int j = 0; j < 3; ++j) { v[j] = in[(size_t)t * HID + tid + 256 * j]; s += v[j]; }
    red[tid] = s; __syncthreads();
    for (int off = 128; off > 0; off >>= 1) {
        if (tid < off) red[tid] += red[tid + off];
        __syncthreads();
    }
    float mu = red[0] * (1.0f / HID);
    __syncthreads();

    float q = 0.0f;
#pragma unroll
    for (int j = 0; j < 3; ++j) { float dd = v[j] - mu; q += dd * dd; }
    red[tid] = q; __syncthreads();
    for (int off = 128; off > 0; off >>= 1) {
        if (tid < off) red[tid] += red[tid + off];
        __syncthreads();
    }
    float rstd = rsqrtf(red[0] * (1.0f / HID) + 1e-5f);
#pragma unroll
    for (int j = 0; j < 3; ++j) {
        int c = tid + 256 * j;
        out[(size_t)t * HID + c] = (v[j] - mu) * rstd * gamma[c] + beta[c];
    }
}

// ---------------------------------------------------------------------------
// Host-side orchestration
// ---------------------------------------------------------------------------
extern "C" void kernel_launch(void* const* d_in, const int* in_sizes, int n_in,
                              void* d_out, int out_size, void* d_ws, size_t ws_size,
                              hipStream_t stream) {
    (void)in_sizes; (void)n_in; (void)out_size; (void)ws_size;

    const float* x      = (const float*)d_in[0];
    const float* W_in   = (const float*)d_in[1];
    const float* conv_w = (const float*)d_in[2];
    const float* conv_b = (const float*)d_in[3];
    const float* W_x    = (const float*)d_in[4];
    const float* W_dt   = (const float*)d_in[5];
    const float* b_dt   = (const float*)d_in[6];
    const float* A_log  = (const float*)d_in[7];
    const float* Dp     = (const float*)d_in[8];
    const float* W_out  = (const float*)d_in[9];
    const float* gamma  = (const float*)d_in[10];
    const float* beta   = (const float*)d_in[11];
    float* out = (float*)d_out;

    // ---- carve workspace (256B aligned slabs), ~141 MB total ----
    char* ws = (char*)d_ws;
    size_t off = 0;
    auto alloc = [&](size_t bytes) -> char* {
        char* p = ws + off;
        off = (off + bytes + 255) & ~(size_t)255;
        return p;
    };
    __bf16* Wb_in   = (__bf16*)alloc((size_t)HID * 2 * INNER * 2);
    __bf16* Wb_conv = (__bf16*)alloc((size_t)KIM * INNER * 2);
    __bf16* Wb_x    = (__bf16*)alloc((size_t)INNER * 2 * STATE * 2);
    __bf16* Wb_dt   = (__bf16*)alloc((size_t)INNER * INNER * 2);
    __bf16* Wb_out  = (__bf16*)alloc((size_t)INNER * HID * 2);
    __bf16* xb      = (__bf16*)alloc((size_t)NTOK * HID * 2);
    float*  xz      = (float*) alloc((size_t)NTOK * 2 * INNER * 4);
    __bf16* im2col  = (__bf16*)alloc((size_t)NTOK * KIM * 2);
    float*  xconv   = (float*) alloc((size_t)NTOK * INNER * 4);
    __bf16* xconvb  = (__bf16*)alloc((size_t)NTOK * INNER * 2);
    float*  BC      = (float*) alloc((size_t)NTOK * 2 * STATE * 4);
    float*  delta   = (float*) alloc((size_t)NTOK * INNER * 4);
    float*  yscan   = (float*) alloc((size_t)NTOK * INNER * 4);
    __bf16* yb      = (__bf16*)alloc((size_t)NTOK * INNER * 2);
    float*  outbuf  = (float*) alloc((size_t)NTOK * HID * 4);

    auto cvt = [&](const float* s, __bf16* d, long n) {
        cvt_f32_bf16<<<dim3((unsigned)((n + 255) / 256)), dim3(256), 0, stream>>>(s, d, n);
    };

    // ---- precision conversion of weights & input ----
    cvt(x,      xb,      (long)NTOK * HID);
    cvt(W_in,   Wb_in,   (long)HID * 2 * INNER);
    cvt(conv_w, Wb_conv, (long)KIM * INNER);     // (K,in,out) flat == [6144,1536]
    cvt(W_x,    Wb_x,    (long)INNER * 2 * STATE);
    cvt(W_dt,   Wb_dt,   (long)INNER * INNER);
    cvt(W_out,  Wb_out,  (long)INNER * HID);

    // ---- in_proj: xz = x @ W_in  [2048,768]x[768,3072] ----
    gemm_bf16_wmma<<<dim3(2 * INNER / 32, NTOK / 64), dim3(256), 0, stream>>>(
        xb, Wb_in, xz, nullptr, nullptr, NTOK, 2 * INNER, HID, 0);

    // ---- im2col for conv ----
    pack_im2col<<<dim3((unsigned)(((long)NTOK * KIM + 255) / 256)), dim3(256), 0, stream>>>(xz, im2col);

    // ---- conv as GEMM + bias + SiLU (f32 + bf16 mirror) ----
    gemm_bf16_wmma<<<dim3(INNER / 64, NTOK / 64), dim3(256), 0, stream>>>(
        im2col, Wb_conv, xconv, xconvb, conv_b, NTOK, INNER, KIM, 1);

    // ---- B/C projection: [2048,1536]x[1536,32] (N<64 -> single col block) ----
    gemm_bf16_wmma<<<dim3(1, NTOK / 64), dim3(256), 0, stream>>>(
        xconvb, Wb_x, BC, nullptr, nullptr, NTOK, 2 * STATE, INNER, 0);

    // ---- delta = softplus(x_conv @ W_dt + b_dt) ----
    gemm_bf16_wmma<<<dim3(INNER / 64, NTOK / 64), dim3(256), 0, stream>>>(
        xconvb, Wb_dt, delta, nullptr, b_dt, NTOK, INNER, INNER, 2);

    // ---- sequential selective scan ----
    scan_kernel<<<dim3(INNER / 256, BSZ), dim3(256), 0, stream>>>(
        xconv, delta, BC, A_log, yscan);

    // ---- skip connection + gating -> bf16 ----
    gate_kernel<<<dim3((unsigned)(((long)NTOK * INNER + 255) / 256)), dim3(256), 0, stream>>>(
        yscan, xconv, xz, Dp, yb);

    // ---- out_proj: [2048,1536]x[1536,768] ----
    gemm_bf16_wmma<<<dim3(HID / 64, NTOK / 64), dim3(256), 0, stream>>>(
        yb, Wb_out, outbuf, nullptr, nullptr, NTOK, HID, INNER, 0);

    // ---- LayerNorm ----
    layernorm_kernel<<<dim3(NTOK), dim3(256), 0, stream>>>(outbuf, gamma, beta, out);
}
